// CrossAttentionModule_31705448579688
// MI455X (gfx1250) — compile-verified, matrix operand order checked by reading, not run on hardware
//
#include <hip/hip_runtime.h>
#include <hip/hip_bf16.h>
#include <math.h>

typedef __bf16 bf16_t;
typedef __attribute__((ext_vector_type(16))) __bf16 v16bf;
typedef __attribute__((ext_vector_type(8)))  __bf16 v8bf;
typedef __attribute__((ext_vector_type(8)))  float  v8f;

#define DMODEL 1024
#define NHEADS 16
#define HDIM   64
#define DFF    4096

static __device__ __forceinline__ bf16_t f2bf(float x) { return (bf16_t)x; }

static __device__ __forceinline__ v8f wmma_bf16(v16bf a, v16bf b, v8f c) {
  // D = A(16x32 bf16) x B(32x16 bf16) + C(16x16 f32)
  return __builtin_amdgcn_wmma_f32_16x16x32_bf16(false, a, false, b, (short)0, c,
                                                 false, false);
}

// 16x32 bf16 operand fragment in the ISA A-layout (also the B operand when the
// source matrix is K-contiguous per output column, i.e. B = W^T, W in [N,K]).
// Lane L (r = L&15, h = L>>4) holds row r, K in {8h..8h+7} U {16+8h..16+8h+7}.
static __device__ __forceinline__ v16bf load_frag(const bf16_t* base, int ld, int lane) {
  const int r = lane & 15, h = lane >> 4;
  const bf16_t* p = base + (size_t)r * ld + 8 * h;
  v8bf lo = *(const v8bf*)p;
  v8bf hi = *(const v8bf*)(p + 16);
  v16bf out;
#pragma unroll
  for (int i = 0; i < 8; ++i) { out[i] = lo[i]; out[i + 8] = hi[i]; }
  return out;
}

// Async global -> LDS 16B copy (CDNA5, tracked by ASYNCcnt).
static __device__ __forceinline__ void async_ld_b128(uint32_t lds_off, const void* gaddr) {
  asm volatile("global_load_async_to_lds_b128 %0, %1, off"
               :: "v"(lds_off), "v"(gaddr) : "memory");
}
static __device__ __forceinline__ void wait_async0() {
  asm volatile("s_wait_asynccnt 0x0" ::: "memory");
}
static __device__ __forceinline__ uint32_t lds_off32(const void* p) {
  return (uint32_t)(uintptr_t)p;   // generic LDS addr: low 32 bits = LDS offset
}

// ---------------------------------------------------------------------------
// LayerNorm (fp32 in) -> bf16 out.  One block per row of DMODEL.
// ---------------------------------------------------------------------------
__global__ __launch_bounds__(256)
void ln_cast_kernel(const float* __restrict__ X, const float* __restrict__ gamma,
                    const float* __restrict__ beta, bf16_t* __restrict__ Y) {
  const int row = blockIdx.x;
  const float* x = X + (size_t)row * DMODEL;
  float s = 0.f, s2 = 0.f;
  for (int i = threadIdx.x; i < DMODEL; i += 256) {
    float v = x[i]; s += v; s2 += v * v;
  }
#pragma unroll
  for (int off = 1; off < 32; off <<= 1) {
    s  += __shfl_xor(s,  off, 32);
    s2 += __shfl_xor(s2, off, 32);
  }
  __shared__ float ws[8], ws2[8];
  if ((threadIdx.x & 31) == 0) { ws[threadIdx.x >> 5] = s; ws2[threadIdx.x >> 5] = s2; }
  __syncthreads();
  s = 0.f; s2 = 0.f;
#pragma unroll
  for (int w = 0; w < 8; ++w) { s += ws[w]; s2 += ws2[w]; }
  const float mu   = s * (1.0f / DMODEL);
  const float var  = s2 * (1.0f / DMODEL) - mu * mu;
  const float rstd = rsqrtf(var + 1e-5f);
  bf16_t* y = Y + (size_t)row * DMODEL;
  for (int i = threadIdx.x; i < DMODEL; i += 256)
    y[i] = f2bf((x[i] - mu) * rstd * gamma[i] + beta[i]);
}

// ---------------------------------------------------------------------------
// fp32 -> bf16 weight cast (n divisible by 1024)
// ---------------------------------------------------------------------------
__global__ __launch_bounds__(256)
void cast_w_kernel(const float* __restrict__ in, bf16_t* __restrict__ out, int n) {
  int i = (blockIdx.x * 256 + threadIdx.x) * 4;
  if (i < n) {
    float4 v = *(const float4*)(in + i);
    out[i + 0] = f2bf(v.x); out[i + 1] = f2bf(v.y);
    out[i + 2] = f2bf(v.z); out[i + 3] = f2bf(v.w);
  }
}

// ---------------------------------------------------------------------------
// Tiled WMMA GEMM:  C[M,N] = A[M,K] @ W[N,K]^T   (both bf16, K-contiguous)
// Block tile 64x256, K-step 32, 8 waves (2x4), each wave 32x64 (8 WMMA/step).
// Double-buffered LDS filled with global_load_async_to_lds_b128 (ASYNCcnt).
// LDS row stride 40 bf16 (80B) -> conflict-free ds_load_b128 fragment reads.
// EPI: 1 = bf16 out + exact GELU
//      2 = f32 out  = residual + acc
//      3 = bf16 out split-head [B,H,Trows,64]   (Q and K)
//      4 = bf16 out split-head transposed [B,H,64,Trows]  (V)
// ---------------------------------------------------------------------------
template <int EPI>
__global__ __launch_bounds__(256)
void gemm_bf16_kernel(const bf16_t* __restrict__ A, const bf16_t* __restrict__ W,
                      const float* __restrict__ R, void* __restrict__ OutV,
                      int M, int N, int K, int Trows) {
  __shared__ bf16_t At[2][64][40];
  __shared__ bf16_t Wt[2][256][40];
  const int tid = threadIdx.x, lane = tid & 31, wave = tid >> 5;
  const int wm = wave >> 2, wn = wave & 3;           // 2 x 4 wave grid
  const int m0 = blockIdx.y * 64, n0 = blockIdx.x * 256;
  v8f acc[2][4] = {};

  // Per-thread staging assignments: A = 256 16B-chunks (1/thread),
  // W = 1024 16B-chunks (4/thread).
  const int arow = tid >> 2, acol = (tid & 3) * 8;
  const bf16_t* gA = A + (size_t)(m0 + arow) * K + acol;

  auto stage = [&](int k0, int buf) {
    async_ld_b128(lds_off32(&At[buf][arow][acol]), gA + k0);
#pragma unroll
    for (int rep = 0; rep < 4; ++rep) {
      const int c = tid + rep * 256;
      const int wrow = c >> 2, wcol = (c & 3) * 8;
      async_ld_b128(lds_off32(&Wt[buf][wrow][wcol]),
                    W + (size_t)(n0 + wrow) * K + k0 + wcol);
    }
  };

  stage(0, 0);
  wait_async0();
  __syncthreads();

  int buf = 0;
  for (int k0 = 0; k0 < K; k0 += 32) {
    const bool more = (k0 + 32 < K);
    if (more) stage(k0 + 32, buf ^ 1);     // overlap next stage with compute

    v16bf a0 = load_frag(&At[buf][wm * 32][0],      40, lane);
    v16bf a1 = load_frag(&At[buf][wm * 32 + 16][0], 40, lane);
    v16bf b0 = load_frag(&Wt[buf][wn * 64][0],      40, lane);
    v16bf b1 = load_frag(&Wt[buf][wn * 64 + 16][0], 40, lane);
    v16bf b2 = load_frag(&Wt[buf][wn * 64 + 32][0], 40, lane);
    v16bf b3 = load_frag(&Wt[buf][wn * 64 + 48][0], 40, lane);
    acc[0][0] = wmma_bf16(a0, b0, acc[0][0]);
    acc[1][0] = wmma_bf16(a1, b0, acc[1][0]);
    acc[0][1] = wmma_bf16(a0, b1, acc[0][1]);
    acc[1][1] = wmma_bf16(a1, b1, acc[1][1]);
    acc[0][2] = wmma_bf16(a0, b2, acc[0][2]);
    acc[1][2] = wmma_bf16(a1, b2, acc[1][2]);
    acc[0][3] = wmma_bf16(a0, b3, acc[0][3]);
    acc[1][3] = wmma_bf16(a1, b3, acc[1][3]);

    if (more) wait_async0();               // own async copies landed
    __syncthreads();                       // everyone done reading buf
    buf ^= 1;
  }

  const int hi = lane >> 4, nn = lane & 15;          // C: row = r + 8*hi, col = nn
#pragma unroll
  for (int i = 0; i < 2; ++i)
#pragma unroll
    for (int j = 0; j < 4; ++j)
#pragma unroll
      for (int r = 0; r < 8; ++r) {
        const int row = m0 + wm * 32 + 16 * i + r + 8 * hi;
        const int col = n0 + wn * 64 + 16 * j + nn;
        const float v = acc[i][j][r];
        if constexpr (EPI == 1) {
          float g = 0.5f * v * (1.0f + erff(v * 0.70710678118f));
          ((bf16_t*)OutV)[(size_t)row * N + col] = f2bf(g);
        } else if constexpr (EPI == 2) {
          ((float*)OutV)[(size_t)row * N + col] = R[(size_t)row * N + col] + v;
        } else if constexpr (EPI == 3) {
          int b = row / Trows, t = row % Trows, h = col >> 6, d = col & 63;
          ((bf16_t*)OutV)[((((size_t)b * NHEADS + h) * Trows + t) << 6) + d] = f2bf(v);
        } else { // EPI == 4 : V stored transposed [B,H,Dh,S]
          int b = row / Trows, sIdx = row % Trows, h = col >> 6, d = col & 63;
          ((bf16_t*)OutV)[(((size_t)b * NHEADS + h) * HDIM + d) * Trows + sIdx] = f2bf(v);
        }
      }
}

// ---------------------------------------------------------------------------
// Flash attention, one wave per (b,h, 16-row Q tile).
// Scores computed TRANSPOSED (S^T = K @ Q^T, identical fragment loads with
// A/B roles swapped) so the softmax reduction over S is a per-lane reduction
// across the 8 accumulator VGPRs + one xor-16 shuffle; m/l is one scalar per
// lane (per output column t).  P is written transposed to LDS with two
// ds_store_b128 per lane, which converts it back to the A-layout for P@V.
// All 8 global fragments (4xK, 4xV) are hoisted to the top of the chunk so
// the loads clause together and stay in flight during the softmax VALU work.
// Q,K: [B,H,T,64] bf16; V: [B,H,64,S] bf16 (transposed); O: [B,T,D] bf16.
// ---------------------------------------------------------------------------
__global__ __launch_bounds__(256)
void flash_attn_kernel(const bf16_t* __restrict__ Q, const bf16_t* __restrict__ Kc,
                       const bf16_t* __restrict__ Vt, bf16_t* __restrict__ O,
                       int T, int S) {
  __shared__ bf16_t Pst[8][16][40];
  const int lane = threadIdx.x & 31, wave = threadIdx.x >> 5;
  const int tile = blockIdx.x * 8 + wave;
  const int tilesPerBH = T >> 4;
  const int bh = tile / tilesPerBH;
  const int t0 = (tile % tilesPerBH) << 4;
  const bf16_t* q = Q  + ((size_t)bh * T + t0) * HDIM;
  const bf16_t* k = Kc + (size_t)bh * S * HDIM;
  const bf16_t* v = Vt + (size_t)bh * HDIM * S;

  const v16bf qf0 = load_frag(q,      HDIM, lane);   // d = 0..31
  const v16bf qf1 = load_frag(q + 32, HDIM, lane);   // d = 32..63

  v8f o0 = {}, o1 = {}, o2 = {}, o3 = {};
  float mCol = -1e30f, lCol = 0.f;                   // per column t = lane&15
  bf16_t (*P)[40] = Pst[wave];
  const int hi = lane >> 4, nn = lane & 15;

  for (int s0 = 0; s0 < S; s0 += 32) {
    const bf16_t* krow = k + (size_t)s0 * HDIM;
    if (s0 + 32 < S) __builtin_prefetch(krow + 32 * HDIM, 0, 1);

    // Hoist all global fragments for this chunk (16 x global_load_b128).
    v16bf kf0 = load_frag(krow,                  HDIM, lane);
    v16bf kf1 = load_frag(krow + 32,             HDIM, lane);
    v16bf kf2 = load_frag(krow + 16 * HDIM,      HDIM, lane);
    v16bf kf3 = load_frag(krow + 16 * HDIM + 32, HDIM, lane);
    v16bf vf0 = load_frag(v + (size_t)0  * S + s0, S, lane);
    v16bf vf1 = load_frag(v + (size_t)16 * S + s0, S, lane);
    v16bf vf2 = load_frag(v + (size_t)32 * S + s0, S, lane);
    v16bf vf3 = load_frag(v + (size_t)48 * S + s0, S, lane);

    // tA[r] = score[s0 + r + 8hi][t0 + nn], tB = same for s0+16..31
    v8f tA = {}, tB = {};
    tA = wmma_bf16(kf0, qf0, tA);
    tA = wmma_bf16(kf1, qf1, tA);
    tB = wmma_bf16(kf2, qf0, tB);
    tB = wmma_bf16(kf3, qf1, tB);

    float mt = -1e30f;
#pragma unroll
    for (int r = 0; r < 8; ++r) {
      tA[r] *= 0.125f; tB[r] *= 0.125f;              // 1/sqrt(64)
      mt = fmaxf(mt, fmaxf(tA[r], tB[r]));
    }
    mt = fmaxf(mt, __shfl_xor(mt, 16, 32));          // combine the two halves
    const float mn   = fmaxf(mCol, mt);
    const float corr = __expf(mCol - mn);
    mCol = mn;

    float ps = 0.f;
    v8bf pA, pB;
#pragma unroll
    for (int r = 0; r < 8; ++r) {
      float pa = __expf(tA[r] - mn), pb = __expf(tB[r] - mn);
      ps += pa + pb;
      pA[r] = f2bf(pa); pB[r] = f2bf(pb);
    }
    ps += __shfl_xor(ps, 16, 32);
    lCol = lCol * corr + ps;

    // transposed store: P[t=nn][sLocal], lane covers sLocal 8hi..8hi+7 (+16)
    *(v8bf*)&P[nn][8 * hi]      = pA;
    *(v8bf*)&P[nn][16 + 8 * hi] = pB;

    // broadcast per-row correction (row t = r + 8hi lives in lane r + 8hi)
#pragma unroll
    for (int r = 0; r < 8; ++r) {
      float cr = __shfl(corr, r + 8 * hi, 32);
      o0[r] *= cr; o1[r] *= cr; o2[r] *= cr; o3[r] *= cr;
    }

    asm volatile("s_wait_dscnt 0x0" ::: "memory");   // same-wave LDS RAW
    v16bf pf = load_frag(&P[0][0], 40, lane);        // A-layout 16x32 (t x s)
    o0 = wmma_bf16(pf, vf0, o0);
    o1 = wmma_bf16(pf, vf1, o1);
    o2 = wmma_bf16(pf, vf2, o2);
    o3 = wmma_bf16(pf, vf3, o3);
  }

  const float invC = 1.0f / lCol;
  const int b = bh >> 4, h = bh & 15;
#pragma unroll
  for (int r = 0; r < 8; ++r) {
    const float ir = __shfl(invC, r + 8 * hi, 32);
    const size_t base = ((size_t)b * T + (t0 + r + 8 * hi)) * DMODEL + h * HDIM + nn;
    O[base +  0] = f2bf(o0[r] * ir);
    O[base + 16] = f2bf(o1[r] * ir);
    O[base + 32] = f2bf(o2[r] * ir);
    O[base + 48] = f2bf(o3[r] * ir);
  }
}

// ---------------------------------------------------------------------------
extern "C" void kernel_launch(void* const* d_in, const int* in_sizes, int n_in,
                              void* d_out, int out_size, void* d_ws, size_t ws_size,
                              hipStream_t stream) {
  (void)in_sizes; (void)n_in; (void)out_size; (void)ws_size;
  const float* x    = (const float*)d_in[0];
  const float* ctx  = (const float*)d_in[1];
  const float* Wq   = (const float*)d_in[2];
  const float* Wk   = (const float*)d_in[3];
  const float* Wv   = (const float*)d_in[4];
  const float* Wo   = (const float*)d_in[5];
  const float* W1   = (const float*)d_in[6];
  const float* W2   = (const float*)d_in[7];
  const float* gam1 = (const float*)d_in[8];
  const float* bet1 = (const float*)d_in[9];
  const float* gamc = (const float*)d_in[10];
  const float* betc = (const float*)d_in[11];
  const float* gam2 = (const float*)d_in[12];
  const float* bet2 = (const float*)d_in[13];

  const int B = 4, T = 2048, S = 2048;
  const int M = B * T;                 // 8192 rows

  char* wp = (char*)d_ws;
  auto take = [&](size_t bytes) -> void* {
    void* p = (void*)wp;
    wp += (bytes + 255) & ~(size_t)255;
    return p;
  };
  bf16_t* xn  = (bf16_t*)take((size_t)M * DMODEL * 2);
  bf16_t* cn  = (bf16_t*)take((size_t)M * DMODEL * 2);
  bf16_t* WqB = (bf16_t*)take((size_t)DMODEL * DMODEL * 2);
  bf16_t* WkB = (bf16_t*)take((size_t)DMODEL * DMODEL * 2);
  bf16_t* WvB = (bf16_t*)take((size_t)DMODEL * DMODEL * 2);
  bf16_t* WoB = (bf16_t*)take((size_t)DMODEL * DMODEL * 2);
  bf16_t* W1B = (bf16_t*)take((size_t)DFF * DMODEL * 2);
  bf16_t* W2B = (bf16_t*)take((size_t)DMODEL * DFF * 2);
  bf16_t* Qb  = (bf16_t*)take((size_t)M * DMODEL * 2);
  bf16_t* Kb  = (bf16_t*)take((size_t)M * DMODEL * 2);
  bf16_t* Vtb = (bf16_t*)take((size_t)M * DMODEL * 2);
  bf16_t* Ob  = (bf16_t*)take((size_t)M * DMODEL * 2);
  float*  x1  = (float*) take((size_t)M * DMODEL * 4);
  bf16_t* hb  = (bf16_t*)take((size_t)M * DMODEL * 2);
  bf16_t* f1b = (bf16_t*)take((size_t)M * DFF * 2);

  dim3 blk(256);

  // LayerNorms -> bf16
  ln_cast_kernel<<<M, blk, 0, stream>>>(x,   gam1, bet1, xn);
  ln_cast_kernel<<<M, blk, 0, stream>>>(ctx, gamc, betc, cn);

  // Weight casts
  cast_w_kernel<<<(DMODEL * DMODEL) / 1024, blk, 0, stream>>>(Wq, WqB, DMODEL * DMODEL);
  cast_w_kernel<<<(DMODEL * DMODEL) / 1024, blk, 0, stream>>>(Wk, WkB, DMODEL * DMODEL);
  cast_w_kernel<<<(DMODEL * DMODEL) / 1024, blk, 0, stream>>>(Wv, WvB, DMODEL * DMODEL);
  cast_w_kernel<<<(DMODEL * DMODEL) / 1024, blk, 0, stream>>>(Wo, WoB, DMODEL * DMODEL);
  cast_w_kernel<<<(DFF * DMODEL) / 1024,    blk, 0, stream>>>(W1, W1B, DFF * DMODEL);
  cast_w_kernel<<<(DMODEL * DFF) / 1024,    blk, 0, stream>>>(W2, W2B, DMODEL * DFF);

  // Q/K/V projections (split-head layouts)
  dim3 gD(DMODEL / 256, M / 64);
  gemm_bf16_kernel<3><<<gD, blk, 0, stream>>>(xn, WqB, nullptr, Qb,  M, DMODEL, DMODEL, T);
  gemm_bf16_kernel<3><<<gD, blk, 0, stream>>>(cn, WkB, nullptr, Kb,  M, DMODEL, DMODEL, S);
  gemm_bf16_kernel<4><<<gD, blk, 0, stream>>>(cn, WvB, nullptr, Vtb, M, DMODEL, DMODEL, S);

  // Fused flash attention
  flash_attn_kernel<<<(B * NHEADS * (T / 16)) / 8, blk, 0, stream>>>(Qb, Kb, Vtb, Ob, T, S);

  // Output projection + residual (fp32)
  gemm_bf16_kernel<2><<<gD, blk, 0, stream>>>(Ob, WoB, x, x1, M, DMODEL, DMODEL, 0);

  // Second LN, FFN with GELU, down-proj + residual -> d_out (fp32)
  ln_cast_kernel<<<M, blk, 0, stream>>>(x1, gam2, bet2, hb);
  dim3 gF(DFF / 256, M / 64);
  gemm_bf16_kernel<1><<<gF, blk, 0, stream>>>(hb, W1B, nullptr, f1b, M, DFF, DMODEL, 0);
  gemm_bf16_kernel<2><<<gD, blk, 0, stream>>>(f1b, W2B, x1, (float*)d_out, M, DMODEL, DFF, 0);
}
